// TsideAtt_40767829574403
// MI455X (gfx1250) — matrix operand order repaired and checked by reading, not compiled, not run on hardware
//
#include <hip/hip_runtime.h>
#include <hip/hip_bf16.h>
#include <math.h>

typedef __bf16 bf16_t;
typedef __attribute__((ext_vector_type(16))) __bf16 v16bf;
typedef __attribute__((ext_vector_type(8)))  __bf16 v8bf;
typedef __attribute__((ext_vector_type(8)))  float  v8f;
typedef __attribute__((ext_vector_type(4)))  float  v4f;

// ---------------------------------------------------------------------------
// Fragment loaders.
// A (16-bit 16x32): lane L row M=L%16; element j -> K = j + (j>=8?8:0) + (L>=16?8:0)
//   => two contiguous 8-elem runs at (base+hiA) and (base+hiA+16), hiA = L>=16 ? 8 : 0
// B (16-bit 32x16): lane L col N=L%16; element j -> K = j + (j>=8?8:0) + (L>=16?8:0)
//   => two contiguous 8-elem runs at (base+hiB) and (base+hiB+16), hiB = L>=16 ? 8 : 0
// ---------------------------------------------------------------------------

static __device__ __forceinline__ v16bf cat8(v8bf lo, v8bf hi) {
  return __builtin_shufflevector(lo, hi, 0,1,2,3,4,5,6,7,8,9,10,11,12,13,14,15);
}

static __device__ __forceinline__ v16bf frag_a_bf16(const bf16_t* p) {
  v8bf lo = *(const v8bf*)(p);        // K run j=0..7
  v8bf hi = *(const v8bf*)(p + 16);   // K run j=8..15
  return cat8(lo, hi);
}

static __device__ __forceinline__ v16bf frag_b_bf16(const bf16_t* p) {
  v8bf lo = *(const v8bf*)(p);
  v8bf hi = *(const v8bf*)(p + 16);
  return cat8(lo, hi);
}

static __device__ __forceinline__ v16bf frag_a_f32(const float* p) {
  v4f a0 = *(const v4f*)(p);
  v4f a1 = *(const v4f*)(p + 4);
  v4f a2 = *(const v4f*)(p + 16);
  v4f a3 = *(const v4f*)(p + 20);
  v16bf r;
  r[0]=(bf16_t)a0[0];  r[1]=(bf16_t)a0[1];  r[2]=(bf16_t)a0[2];  r[3]=(bf16_t)a0[3];
  r[4]=(bf16_t)a1[0];  r[5]=(bf16_t)a1[1];  r[6]=(bf16_t)a1[2];  r[7]=(bf16_t)a1[3];
  r[8]=(bf16_t)a2[0];  r[9]=(bf16_t)a2[1];  r[10]=(bf16_t)a2[2]; r[11]=(bf16_t)a2[3];
  r[12]=(bf16_t)a3[0]; r[13]=(bf16_t)a3[1]; r[14]=(bf16_t)a3[2]; r[15]=(bf16_t)a3[3];
  return r;
}

static __device__ __forceinline__ v16bf frag_b_f32(const float* p) {
  v4f a0 = *(const v4f*)(p);
  v4f a1 = *(const v4f*)(p + 4);
  v4f a2 = *(const v4f*)(p + 16);
  v4f a3 = *(const v4f*)(p + 20);
  v16bf r;
  r[0]=(bf16_t)a0[0];  r[1]=(bf16_t)a0[1];  r[2]=(bf16_t)a0[2];  r[3]=(bf16_t)a0[3];
  r[4]=(bf16_t)a1[0];  r[5]=(bf16_t)a1[1];  r[6]=(bf16_t)a1[2];  r[7]=(bf16_t)a1[3];
  r[8]=(bf16_t)a2[0];  r[9]=(bf16_t)a2[1];  r[10]=(bf16_t)a2[2]; r[11]=(bf16_t)a2[3];
  r[12]=(bf16_t)a3[0]; r[13]=(bf16_t)a3[1]; r[14]=(bf16_t)a3[2]; r[15]=(bf16_t)a3[3];
  return r;
}

static __device__ __forceinline__ v8f zero8() {
  v8f z = {0.f,0.f,0.f,0.f,0.f,0.f,0.f,0.f};
  return z;
}

#define WMMA_BF16(A, B, C) \
  __builtin_amdgcn_wmma_f32_16x16x32_bf16(false, (A), false, (B), (short)0, (C), false, false)

// ---------------------------------------------------------------------------
// One k-step (K += 32) of the 16x128 per-wave GEMM.
// Literal buffer indices only: consume Bf[t&3], prefetch Bf[(t+2)&3] (dist 2),
// A double-buffered on literal CA; PF (literal) gates cross-step prefetch.
// ---------------------------------------------------------------------------
#define GEMM_STEP(CA, K0, KN, PF, FRAG_A, FRAG_B, AROW, BROW, TSTR)               \
  do {                                                                            \
    Bf[2] = FRAG_B((BROW) + (size_t)2 * (TSTR) + (K0));                           \
    acc[0] = WMMA_BF16(Af[CA], Bf[0], acc[0]);                                    \
    Bf[3] = FRAG_B((BROW) + (size_t)3 * (TSTR) + (K0));                           \
    acc[1] = WMMA_BF16(Af[CA], Bf[1], acc[1]);                                    \
    Bf[0] = FRAG_B((BROW) + (size_t)4 * (TSTR) + (K0));                           \
    acc[2] = WMMA_BF16(Af[CA], Bf[2], acc[2]);                                    \
    Bf[1] = FRAG_B((BROW) + (size_t)5 * (TSTR) + (K0));                           \
    if (PF) Af[(CA) ^ 1] = FRAG_A((AROW) + (KN));                                 \
    acc[3] = WMMA_BF16(Af[CA], Bf[3], acc[3]);                                    \
    Bf[2] = FRAG_B((BROW) + (size_t)6 * (TSTR) + (K0));                           \
    acc[4] = WMMA_BF16(Af[CA], Bf[0], acc[4]);                                    \
    Bf[3] = FRAG_B((BROW) + (size_t)7 * (TSTR) + (K0));                           \
    acc[5] = WMMA_BF16(Af[CA], Bf[1], acc[5]);                                    \
    if (PF) Bf[0] = FRAG_B((BROW) + (KN));                                        \
    acc[6] = WMMA_BF16(Af[CA], Bf[2], acc[6]);                                    \
    if (PF) Bf[1] = FRAG_B((BROW) + (size_t)1 * (TSTR) + (KN));                   \
    acc[7] = WMMA_BF16(Af[CA], Bf[3], acc[7]);                                    \
  } while (0)

// ---------------------------------------------------------------------------
// Kernel 1: projection.  out_bf16[M,512] = X_f32[M,512] @ W_f32[512,512]^T
// Grid: (M/128, 4). 8 waves; wave w -> rows [bx*128+16w,+16), cols [by*128,+128).
// ---------------------------------------------------------------------------
__global__ __launch_bounds__(256) void proj_kernel(const float* __restrict__ X,
                                                   const float* __restrict__ W,
                                                   bf16_t* __restrict__ out) {
  const int lane = threadIdx.x & 31;
  const int wave = threadIdx.x >> 5;
  const int l16  = lane & 15;
  const int row0 = blockIdx.x * 128 + wave * 16;
  const int col0 = blockIdx.y * 128;
  const int hiA  = (lane >= 16) ? 8 : 0;
  const int hiB  = (lane >= 16) ? 8 : 0;

  const float* arow = X + (size_t)(row0 + l16) * 512 + hiA;
  const float* brow = W + (size_t)(col0 + l16) * 512 + hiB;   // + t*16*512 per N-tile

  v8f acc[8];
#pragma unroll
  for (int t = 0; t < 8; ++t) acc[t] = zero8();

  v16bf Af[2], Bf[4];
  Af[0] = frag_a_f32(arow);
  Bf[0] = frag_b_f32(brow);
  Bf[1] = frag_b_f32(brow + (size_t)(16 * 512));

  for (int k0 = 0; k0 < 512 - 64; k0 += 64) {
    GEMM_STEP(0, k0,      k0 + 32, true, frag_a_f32, frag_b_f32, arow, brow, 16 * 512);
    GEMM_STEP(1, k0 + 32, k0 + 64, true, frag_a_f32, frag_b_f32, arow, brow, 16 * 512);
  }
  GEMM_STEP(0, 512 - 64, 512 - 32, true,  frag_a_f32, frag_b_f32, arow, brow, 16 * 512);
  GEMM_STEP(1, 512 - 32, 512,      false, frag_a_f32, frag_b_f32, arow, brow, 16 * 512);

  const int rb = row0 + ((lane >= 16) ? 8 : 0);   // D: VGPR g -> M = g (+8 hi lanes)
#pragma unroll
  for (int t = 0; t < 8; ++t)
#pragma unroll
    for (int g = 0; g < 8; ++g)
      out[(size_t)(rb + g) * 512 + col0 + t * 16 + l16] = (bf16_t)acc[t][g];
}

// ---------------------------------------------------------------------------
// Kernel 2: V transpose + f32->bf16.  Vt[b][v][k] = V[b][k][v]
// ---------------------------------------------------------------------------
__global__ __launch_bounds__(256) void transpose_v(const float* __restrict__ V,
                                                   bf16_t* __restrict__ Vt) {
  __shared__ bf16_t tile[32][33];
  const int j  = threadIdx.x & 31;
  const int i0 = threadIdx.x >> 5;            // 0..7
  const int v0 = blockIdx.x * 32;
  const int k0 = blockIdx.y * 32;
  const int b  = blockIdx.z;
#pragma unroll
  for (int ii = 0; ii < 4; ++ii) {
    int kk = i0 + ii * 8;
    tile[j][kk] = (bf16_t)V[((size_t)b * 1024 + k0 + kk) * 512 + v0 + j];
  }
  __syncthreads();
#pragma unroll
  for (int ii = 0; ii < 4; ++ii) {
    int vv = i0 + ii * 8;
    Vt[((size_t)b * 512 + v0 + vv) * 1024 + k0 + j] = tile[vv][j];
  }
}

// ---------------------------------------------------------------------------
// Kernel 3: fused logits + masked-exp L1 "softmax".
// One WG per (b, 16-row q tile). 8 waves x 128 cols cover Tk=1024.
// logits -> 64KB LDS -> row reductions via wave32 shfl_xor -> w fp32 to d_out
// (+ optional bf16 copy to workspace for the out-GEMM's A operand).
// ---------------------------------------------------------------------------
__global__ __launch_bounds__(256) void scores_kernel(const bf16_t* __restrict__ pq,
                                                     const bf16_t* __restrict__ pk,
                                                     const int* __restrict__ tlen,
                                                     const int* __restrict__ slen,
                                                     float* __restrict__ wout,
                                                     bf16_t* __restrict__ wbf,
                                                     int write_bf) {
  __shared__ float slog[16 * 1024];           // 64 KB
  const int lane = threadIdx.x & 31;
  const int wave = threadIdx.x >> 5;
  const int l16  = lane & 15;
  const int b    = blockIdx.y;
  const int q0   = blockIdx.x * 16;
  const int hiA  = (lane >= 16) ? 8 : 0;
  const int hiB  = (lane >= 16) ? 8 : 0;

  const bf16_t* arow = pq + ((size_t)b * 512 + q0 + l16) * 512 + hiA;
  const bf16_t* brow = pk + ((size_t)b * 1024 + wave * 128 + l16) * 512 + hiB;

  v8f acc[8];
#pragma unroll
  for (int t = 0; t < 8; ++t) acc[t] = zero8();

  v16bf Af[2], Bf[4];
  Af[0] = frag_a_bf16(arow);
  Bf[0] = frag_b_bf16(brow);
  Bf[1] = frag_b_bf16(brow + (size_t)(16 * 512));

  for (int k0 = 0; k0 < 512 - 64; k0 += 64) {
    GEMM_STEP(0, k0,      k0 + 32, true, frag_a_bf16, frag_b_bf16, arow, brow, 16 * 512);
    GEMM_STEP(1, k0 + 32, k0 + 64, true, frag_a_bf16, frag_b_bf16, arow, brow, 16 * 512);
  }
  GEMM_STEP(0, 512 - 64, 512 - 32, true,  frag_a_bf16, frag_b_bf16, arow, brow, 16 * 512);
  GEMM_STEP(1, 512 - 32, 512,      false, frag_a_bf16, frag_b_bf16, arow, brow, 16 * 512);

  const int rb = (lane >= 16) ? 8 : 0;
#pragma unroll
  for (int t = 0; t < 8; ++t)
#pragma unroll
    for (int g = 0; g < 8; ++g)
      slog[(rb + g) * 1024 + wave * 128 + t * 16 + l16] = acc[t][g];
  __syncthreads();

  // 16 threads per q-row, each scanning 64 strided cols.
  const int tid = threadIdx.x;
  const int row = tid >> 4;
  const int sub = tid & 15;
  float* rowp = slog + row * 1024;

  float m = -3.0e38f;
#pragma unroll 8
  for (int j = 0; j < 64; ++j) m = fmaxf(m, rowp[sub + 16 * j]);
#pragma unroll
  for (int off = 8; off >= 1; off >>= 1) m = fmaxf(m, __shfl_xor(m, off, 32));

  const int olen = slen[b] / 2;               // spec_lengths // R_FACTOR
  float s = 0.f;
#pragma unroll 8
  for (int j = 0; j < 64; ++j) {
    int col  = sub + 16 * j;
    float e  = __expf(rowp[col] - m);
    e = (col < olen) ? e : 0.f;
    rowp[col] = e;                            // each thread owns its slots
    s += e;
  }
#pragma unroll
  for (int off = 8; off >= 1; off >>= 1) s += __shfl_xor(s, off, 32);

  const float inv = ((q0 + row) < tlen[b]) ? (1.f / fmaxf(s, 1e-12f)) : 0.f;
  const size_t rbase = ((size_t)b * 512 + q0 + row) * 1024;
  float*  wrow = wout + rbase;
  if (write_bf) {
    bf16_t* brow2 = wbf + rbase;
#pragma unroll 8
    for (int j = 0; j < 64; ++j) {
      int col = sub + 16 * j;
      float val = rowp[col] * inv;
      wrow[col]  = val;
      brow2[col] = (bf16_t)val;
    }
  } else {
#pragma unroll 8
    for (int j = 0; j < 64; ++j) {
      int col = sub + 16 * j;
      wrow[col] = rowp[col] * inv;
    }
  }
}

// ---------------------------------------------------------------------------
// Kernel 4a: out[b] = w_bf16[b] @ Vt[b]^T (bf16 x bf16). Preferred path.
// Per batch: M=512, N=512, K=1024. Grid: (4, 4, 32).
// ---------------------------------------------------------------------------
__global__ __launch_bounds__(256) void outmm_bf16_kernel(const bf16_t* __restrict__ wbf,
                                                         const bf16_t* __restrict__ vt,
                                                         float* __restrict__ outp) {
  const int lane = threadIdx.x & 31;
  const int wave = threadIdx.x >> 5;
  const int l16  = lane & 15;
  const int bz   = blockIdx.z;
  const int row0 = blockIdx.x * 128 + wave * 16;
  const int col0 = blockIdx.y * 128;
  const int hiA  = (lane >= 16) ? 8 : 0;
  const int hiB  = (lane >= 16) ? 8 : 0;

  const bf16_t* arow = wbf + ((size_t)bz * 512 + row0 + l16) * 1024 + hiA;
  const bf16_t* brow = vt + ((size_t)bz * 512 + col0 + l16) * 1024 + hiB;

  v8f acc[8];
#pragma unroll
  for (int t = 0; t < 8; ++t) acc[t] = zero8();

  v16bf Af[2], Bf[4];
  Af[0] = frag_a_bf16(arow);
  Bf[0] = frag_b_bf16(brow);
  Bf[1] = frag_b_bf16(brow + (size_t)(16 * 1024));

  for (int k0 = 0; k0 < 1024 - 64; k0 += 64) {
    GEMM_STEP(0, k0,      k0 + 32, true, frag_a_bf16, frag_b_bf16, arow, brow, 16 * 1024);
    GEMM_STEP(1, k0 + 32, k0 + 64, true, frag_a_bf16, frag_b_bf16, arow, brow, 16 * 1024);
  }
  GEMM_STEP(0, 1024 - 64, 1024 - 32, true,  frag_a_bf16, frag_b_bf16, arow, brow, 16 * 1024);
  GEMM_STEP(1, 1024 - 32, 1024,      false, frag_a_bf16, frag_b_bf16, arow, brow, 16 * 1024);

  const int rb = row0 + ((lane >= 16) ? 8 : 0);
#pragma unroll
  for (int t = 0; t < 8; ++t)
#pragma unroll
    for (int g = 0; g < 8; ++g)
      outp[((size_t)bz * 512 + rb + g) * 512 + col0 + t * 16 + l16] = acc[t][g];
}

// ---------------------------------------------------------------------------
// Kernel 4b: out[b] = w[b] (f32, cvt on the fly) @ Vt[b]^T (bf16). Fallback.
// ---------------------------------------------------------------------------
__global__ __launch_bounds__(256) void outmm_f32_kernel(const float* __restrict__ wsrc,
                                                        const bf16_t* __restrict__ vt,
                                                        float* __restrict__ outp) {
  const int lane = threadIdx.x & 31;
  const int wave = threadIdx.x >> 5;
  const int l16  = lane & 15;
  const int bz   = blockIdx.z;
  const int row0 = blockIdx.x * 128 + wave * 16;
  const int col0 = blockIdx.y * 128;
  const int hiA  = (lane >= 16) ? 8 : 0;
  const int hiB  = (lane >= 16) ? 8 : 0;

  const float*  arow = wsrc + ((size_t)bz * 512 + row0 + l16) * 1024 + hiA;
  const bf16_t* brow = vt + ((size_t)bz * 512 + col0 + l16) * 1024 + hiB;

  v8f acc[8];
#pragma unroll
  for (int t = 0; t < 8; ++t) acc[t] = zero8();

  v16bf Af[2], Bf[4];
  Af[0] = frag_a_f32(arow);
  Bf[0] = frag_b_bf16(brow);
  Bf[1] = frag_b_bf16(brow + (size_t)(16 * 1024));

  for (int k0 = 0; k0 < 1024 - 64; k0 += 64) {
    GEMM_STEP(0, k0,      k0 + 32, true, frag_a_f32, frag_b_bf16, arow, brow, 16 * 1024);
    GEMM_STEP(1, k0 + 32, k0 + 64, true, frag_a_f32, frag_b_bf16, arow, brow, 16 * 1024);
  }
  GEMM_STEP(0, 1024 - 64, 1024 - 32, true,  frag_a_f32, frag_b_bf16, arow, brow, 16 * 1024);
  GEMM_STEP(1, 1024 - 32, 1024,      false, frag_a_f32, frag_b_bf16, arow, brow, 16 * 1024);

  const int rb = row0 + ((lane >= 16) ? 8 : 0);
#pragma unroll
  for (int t = 0; t < 8; ++t)
#pragma unroll
    for (int g = 0; g < 8; ++g)
      outp[((size_t)bz * 512 + rb + g) * 512 + col0 + t * 16 + l16] = acc[t][g];
}

// ---------------------------------------------------------------------------
extern "C" void kernel_launch(void* const* d_in, const int* in_sizes, int n_in,
                              void* d_out, int out_size, void* d_ws, size_t ws_size,
                              hipStream_t stream) {
  const float* q    = (const float*)d_in[0];   // [32,512,512]
  const float* kk   = (const float*)d_in[1];   // [32,1024,512]
  const float* v    = (const float*)d_in[2];   // [32,1024,512]
  const float* Wq   = (const float*)d_in[3];   // [512,512]
  const float* Wk   = (const float*)d_in[4];   // [512,512]
  const int*   tlen = (const int*)d_in[5];     // [32]
  const int*   slen = (const int*)d_in[6];     // [32]

  float* out  = (float*)d_out;                           // [32,512,512]
  float* wout = out + (size_t)32 * 512 * 512;            // [32,512,1024]

  const size_t n_pq = (size_t)32 * 512 * 512;            // elements
  const size_t n_pk = (size_t)32 * 1024 * 512;
  const size_t n_vt = (size_t)32 * 512 * 1024;
  const size_t n_w  = (size_t)32 * 512 * 1024;

  bf16_t* pq  = (bf16_t*)d_ws;                           // 16.8 MB
  bf16_t* pk  = pq + n_pq;                               // 33.6 MB
  bf16_t* vt  = pk + n_pk;                               // 33.6 MB
  bf16_t* wbf = vt + n_vt;                               // 33.6 MB (optional)

  const size_t need_bf = (n_pq + n_pk + n_vt + n_w) * sizeof(bf16_t);
  const int use_wbf = (ws_size >= need_bf) ? 1 : 0;      // deterministic per run

  proj_kernel<<<dim3(128, 4), 256, 0, stream>>>(q,  Wq, pq);   // M = 32*512
  proj_kernel<<<dim3(256, 4), 256, 0, stream>>>(kk, Wk, pk);   // M = 32*1024
  transpose_v<<<dim3(16, 32, 32), 256, 0, stream>>>(v, vt);
  scores_kernel<<<dim3(32, 32), 256, 0, stream>>>(pq, pk, tlen, slen, wout, wbf, use_wbf);
  if (use_wbf)
    outmm_bf16_kernel<<<dim3(4, 4, 32), 256, 0, stream>>>(wbf, vt, out);
  else
    outmm_f32_kernel<<<dim3(4, 4, 32), 256, 0, stream>>>(wout, vt, out);
}
